// StackedRNN_34522947125584
// MI455X (gfx1250) — matrix operand order repaired and checked by reading, not compiled, not run on hardware
//
#include <hip/hip_runtime.h>
#include <math.h>

typedef __attribute__((ext_vector_type(16))) _Float16 v16h;
typedef __attribute__((ext_vector_type(8)))  float    v8f;

#define VV 32000
#define EE 512
#define HH 1024
#define BB 32
#define SS 128
#define RR (BB*SS)     // 4096 rows
#define NWG 16         // workgroups in persistent RNN kernel

__device__ __forceinline__ v8f wmma16(v16h a, v16h b, v8f c) {
  return __builtin_amdgcn_wmma_f32_16x16x32_f16(
      /*neg_a=*/false, a, /*neg_b=*/false, b,
      /*c_mod=*/(short)0, c, /*reuse_a=*/false, /*reuse_b=*/false);
}

// Load a 16x32 f16 A-tile from a row-major f16 matrix into WMMA A layout.
// lanes 0-15: row M = m0+lane, K = {0..7, 16..23}+k0
// lanes16-31: row M = m0+lane-16, K = {8..15, 24..31}+k0
__device__ __forceinline__ v16h load_a_rm(const _Float16* __restrict__ A,
                                          int lda, int m0, int k0, int lane) {
  const _Float16* p = A + (size_t)(m0 + (lane & 15)) * lda + k0 + ((lane >> 4) << 3);
  v16h a;
#pragma unroll
  for (int i = 0; i < 8; ++i) a[i] = p[i];
#pragma unroll
  for (int i = 0; i < 8; ++i) a[8 + i] = p[16 + i];
  return a;
}

// ---------------- pack kernels ----------------
// B-tile native layout: tile = 32(K) x 16(N); 512 halves contiguous.
// lane l holds N = n0+(l&15); K = k0 + (l>>4)*8 + i (i<8), k0 + 16 + (l>>4)*8 + (i-8) (i>=8)
__global__ void pack_b_f16(const float* __restrict__ src, _Float16* __restrict__ dst,
                           int K, int N) {
  int wave = threadIdx.x >> 5, lane = threadIdx.x & 31;
  int ktiles = K >> 5, ntiles = N >> 4;
  int t = blockIdx.x * 8 + wave;             // tile index = nt*ktiles + kt
  if (t >= ntiles * ktiles) return;
  int nt = t / ktiles, kt = t - nt * ktiles;
  int n  = nt * 16 + (lane & 15);
  int kb = kt * 32 + ((lane >> 4) << 3);
  v16h v;
#pragma unroll
  for (int i = 0; i < 16; ++i)
    v[i] = (_Float16)src[(size_t)(kb + i + ((i >> 3) << 3)) * N + n];
  *(v16h*)(dst + (size_t)t * 512 + lane * 16) = v;
}

// Embedding gather fused with A-tile pack: X[r,:] = emb[tokens[r]], r = b*S+s.
// A-tile native layout, tile index = mt*(E/32) + kt, 512 halves contiguous.
__global__ void pack_x_f16(const int* __restrict__ tokens, const float* __restrict__ emb,
                           _Float16* __restrict__ dst) {
  int wave = threadIdx.x >> 5, lane = threadIdx.x & 31;
  int t  = blockIdx.x * 8 + wave;            // 4096 tiles total
  int mt = t >> 4, kt = t & 15;
  int r  = mt * 16 + (lane & 15);
  int tok = tokens[r];
  const float* base = emb + (size_t)tok * EE + kt * 32;
  int kb = (lane >> 4) << 3;
  v16h v;
#pragma unroll
  for (int i = 0; i < 8; ++i) v[i]     = (_Float16)base[kb + i];
#pragma unroll
  for (int i = 0; i < 8; ++i) v[8 + i] = (_Float16)base[16 + kb + i];
  *(v16h*)(dst + (size_t)t * 512 + lane * 16) = v;
}

// ---------------- XW0 = X @ W0 + b0 (all timesteps at once) ----------------
__global__ void gemm_xw0(const _Float16* __restrict__ pX, const _Float16* __restrict__ pW0,
                         const float* __restrict__ b0, float* __restrict__ XW0) {
  int wave = threadIdx.x >> 5, lane = threadIdx.x & 31;
  int nt = blockIdx.x * 8 + wave;            // 0..63
  int mt = blockIdx.y;                       // 0..255
  const int ktiles = EE / 32;                // 16
  const _Float16* abase = pX  + (size_t)mt * ktiles * 512;
  const _Float16* bbase = pW0 + (size_t)nt * ktiles * 512;
  v8f acc = {};
  for (int kt = 0; kt < ktiles; ++kt) {
    v16h a = *(const v16h*)(abase + (size_t)kt * 512 + lane * 16);
    v16h b = *(const v16h*)(bbase + (size_t)kt * 512 + lane * 16);
    acc = wmma16(a, b, acc);
  }
  int col  = nt * 16 + (lane & 15);
  int half = lane >> 4;
  float bias = b0[col];
#pragma unroll
  for (int v = 0; v < 8; ++v) {
    int row = mt * 16 + half * 8 + v;
    XW0[(size_t)row * HH + col] = acc[v] + bias;
  }
}

// ---------------- persistent recurrent kernel ----------------
__global__ void rnn_init(_Float16* h0, _Float16* h1, unsigned* bar) {
  int i = blockIdx.x * blockDim.x + threadIdx.x;
  if (i < BB * HH) { h0[i] = (_Float16)0.f; h1[i] = (_Float16)0.f; }
  if (i < 64) bar[i] = 0u;
}

__device__ __forceinline__ void grid_barrier(unsigned* bar, unsigned* localgen) {
  __threadfence();
  __syncthreads();
  if (threadIdx.x == 0) {
    unsigned target = ++(*localgen);
    unsigned old = atomicAdd(&bar[0], 1u);   // monotonically increasing arrival count
    if (old == target * NWG - 1u) {
      atomicAdd(&bar[1], 1u);                // bump generation
    } else {
      while (atomicAdd(&bar[1], 0u) < target) __builtin_amdgcn_s_sleep(2);
    }
  }
  __syncthreads();
  __threadfence();
}

__global__ void rnn_steps(const float* __restrict__ XW0,
                          const _Float16* __restrict__ pU0,
                          const _Float16* __restrict__ pW1,
                          const _Float16* __restrict__ pU1,
                          const float* __restrict__ b1v,
                          _Float16* h0a, _Float16* h0b,
                          _Float16* h1a, _Float16* h1b,
                          _Float16* __restrict__ Y, unsigned* bar) {
  int wave = threadIdx.x >> 5, lane = threadIdx.x & 31;
  int gw = blockIdx.x * 8 + wave;            // 0..127 waves = 2x64 output tiles
  int mt = gw >> 6, nt = gw & 63;
  int m0 = mt * 16;
  int col  = nt * 16 + (lane & 15);
  int half = lane >> 4;
  unsigned localgen = 0;
  const int ktiles = HH / 32;                // 32
  const _Float16* bU0 = pU0 + (size_t)nt * ktiles * 512;
  const _Float16* bW1 = pW1 + (size_t)nt * ktiles * 512;
  const _Float16* bU1 = pU1 + (size_t)nt * ktiles * 512;
  float bias1 = b1v[col];

  for (int s = 0; s < SS; ++s) {
    _Float16* h0p = (s & 1) ? h0b : h0a;
    _Float16* h0n = (s & 1) ? h0a : h0b;
    _Float16* h1p = (s & 1) ? h1b : h1a;
    _Float16* h1n = (s & 1) ? h1a : h1b;

    // phase A: h0n = tanh(XW0[:,s,:] + h0p @ U0)
    v8f acc = {};
    for (int kt = 0; kt < ktiles; ++kt) {
      v16h a = load_a_rm(h0p, HH, m0, kt * 32, lane);
      v16h b = *(const v16h*)(bU0 + (size_t)kt * 512 + lane * 16);
      acc = wmma16(a, b, acc);
    }
#pragma unroll
    for (int v = 0; v < 8; ++v) {
      int bi = m0 + half * 8 + v;            // batch row
      float t = acc[v] + XW0[((size_t)bi * SS + s) * HH + col];
      h0n[bi * HH + col] = (_Float16)tanhf(t);
    }
    grid_barrier(bar, &localgen);

    // phase B: h1n = tanh(h0n @ W1 + h1p @ U1 + b1)
    v8f acc2 = {};
    for (int kt = 0; kt < ktiles; ++kt) {
      v16h a = load_a_rm(h0n, HH, m0, kt * 32, lane);
      v16h b = *(const v16h*)(bW1 + (size_t)kt * 512 + lane * 16);
      acc2 = wmma16(a, b, acc2);
    }
    for (int kt = 0; kt < ktiles; ++kt) {
      v16h a = load_a_rm(h1p, HH, m0, kt * 32, lane);
      v16h b = *(const v16h*)(bU1 + (size_t)kt * 512 + lane * 16);
      acc2 = wmma16(a, b, acc2);
    }
#pragma unroll
    for (int v = 0; v < 8; ++v) {
      int bi = m0 + half * 8 + v;
      float t = tanhf(acc2[v] + bias1);
      h1n[bi * HH + col] = (_Float16)t;
      Y[((size_t)bi * SS + s) * HH + col] = (_Float16)t;
    }
    grid_barrier(bar, &localgen);
  }
}

// ---------------- logits = Y @ Wd + bd  (dominant GEMM) ----------------
// 2x2 register blocking: each wave computes a 32(M) x 32(N) block = 4 WMMA tiles.
// Per k-step: 2 A-tiles + 2 B-tiles feed 4 wmmas (halves operand traffic vs 1x1).
__global__ void gemm_out(const _Float16* __restrict__ Y, const _Float16* __restrict__ pWd,
                         const float* __restrict__ bd, float* __restrict__ out) {
  int wave = threadIdx.x >> 5, lane = threadIdx.x & 31;
  int ntp = blockIdx.x * 8 + wave;           // N tile-pair: 0..999
  int mtp = blockIdx.y;                      // M tile-pair: 0..127
  const int ktiles = HH / 32;                // 32
  int nt0 = ntp * 2, nt1 = nt0 + 1;
  int m0  = mtp * 32;
  const _Float16* bbase0 = pWd + (size_t)nt0 * ktiles * 512;
  const _Float16* bbase1 = pWd + (size_t)nt1 * ktiles * 512;
  v8f acc00 = {}, acc01 = {}, acc10 = {}, acc11 = {};
  for (int kt = 0; kt < ktiles; ++kt) {
    if (kt + 4 < ktiles) {
      __builtin_prefetch(bbase0 + (size_t)(kt + 4) * 512 + lane * 16, 0, 1);
      __builtin_prefetch(bbase1 + (size_t)(kt + 4) * 512 + lane * 16, 0, 1);
    }
    v16h a0 = load_a_rm(Y, HH, m0,      kt * 32, lane);
    v16h a1 = load_a_rm(Y, HH, m0 + 16, kt * 32, lane);
    v16h b0 = *(const v16h*)(bbase0 + (size_t)kt * 512 + lane * 16);
    v16h b1 = *(const v16h*)(bbase1 + (size_t)kt * 512 + lane * 16);
    acc00 = wmma16(a0, b0, acc00);
    acc01 = wmma16(a0, b1, acc01);
    acc10 = wmma16(a1, b0, acc10);
    acc11 = wmma16(a1, b1, acc11);
  }
  int c0   = nt0 * 16 + (lane & 15);
  int c1   = c0 + 16;
  int half = lane >> 4;
  float bias0 = bd[c0];
  float bias1 = bd[c1];
#pragma unroll
  for (int v = 0; v < 8; ++v) {
    int r0 = m0 + half * 8 + v;
    int r1 = r0 + 16;
    out[(size_t)r0 * VV + c0] = acc00[v] + bias0;
    out[(size_t)r0 * VV + c1] = acc01[v] + bias1;
    out[(size_t)r1 * VV + c0] = acc10[v] + bias0;
    out[(size_t)r1 * VV + c1] = acc11[v] + bias1;
  }
}

extern "C" void kernel_launch(void* const* d_in, const int* in_sizes, int n_in,
                              void* d_out, int out_size, void* d_ws, size_t ws_size,
                              hipStream_t stream) {
  const int*   tokens = (const int*)d_in[0];
  const float* emb = (const float*)d_in[1];
  const float* W0  = (const float*)d_in[2];
  const float* U0  = (const float*)d_in[3];
  const float* b0  = (const float*)d_in[4];
  const float* W1  = (const float*)d_in[5];
  const float* U1  = (const float*)d_in[6];
  const float* b1  = (const float*)d_in[7];
  const float* Wd  = (const float*)d_in[8];
  const float* bd  = (const float*)d_in[9];
  float* out = (float*)d_out;

  char* ws = (char*)d_ws;
  size_t off = 0;
  auto wsalloc = [&](size_t bytes) -> char* {
    char* p = ws + off;
    off = (off + bytes + 255) & ~(size_t)255;
    return p;
  };
  _Float16* pX  = (_Float16*)wsalloc((size_t)RR * EE * 2);   //  4.2 MB
  _Float16* pW0 = (_Float16*)wsalloc((size_t)EE * HH * 2);   //  1.0 MB
  _Float16* pU0 = (_Float16*)wsalloc((size_t)HH * HH * 2);   //  2.1 MB
  _Float16* pW1 = (_Float16*)wsalloc((size_t)HH * HH * 2);
  _Float16* pU1 = (_Float16*)wsalloc((size_t)HH * HH * 2);
  _Float16* pWd = (_Float16*)wsalloc((size_t)HH * VV * 2);   // 65.5 MB
  float*    XW0 = (float*)   wsalloc((size_t)RR * HH * 4);   // 16.8 MB
  _Float16* Yb  = (_Float16*)wsalloc((size_t)RR * HH * 2);   //  8.4 MB
  _Float16* h0a = (_Float16*)wsalloc((size_t)BB * HH * 2);
  _Float16* h0b = (_Float16*)wsalloc((size_t)BB * HH * 2);
  _Float16* h1a = (_Float16*)wsalloc((size_t)BB * HH * 2);
  _Float16* h1b = (_Float16*)wsalloc((size_t)BB * HH * 2);
  unsigned* bar = (unsigned*)wsalloc(256);

  pack_x_f16<<<512, 256, 0, stream>>>(tokens, emb, pX);            // 4096 tiles
  pack_b_f16<<<128, 256, 0, stream>>>(W0, pW0, EE, HH);            // 1024 tiles
  pack_b_f16<<<256, 256, 0, stream>>>(U0, pU0, HH, HH);            // 2048 tiles
  pack_b_f16<<<256, 256, 0, stream>>>(W1, pW1, HH, HH);
  pack_b_f16<<<256, 256, 0, stream>>>(U1, pU1, HH, HH);
  pack_b_f16<<<8000, 256, 0, stream>>>(Wd, pWd, HH, VV);           // 64000 tiles
  gemm_xw0<<<dim3(8, 256), 256, 0, stream>>>(pX, pW0, b0, XW0);
  rnn_init<<<128, 256, 0, stream>>>(h0a, h1a, bar);
  rnn_steps<<<NWG, 256, 0, stream>>>(XW0, pU0, pW1, pU1, b1,
                                     h0a, h0b, h1a, h1b, Yb, bar);
  gemm_out<<<dim3(125, 128), 256, 0, stream>>>(Yb, pWd, bd, out);
}
